// LNN_82935818486281
// MI455X (gfx1250) — matrix operand order repaired and checked by reading, not compiled, not run on hardware
//
#include <hip/hip_runtime.h>
#include <cstdint>

typedef __attribute__((ext_vector_type(16))) _Float16 v16h;
typedef __attribute__((ext_vector_type(8)))  float    v8f;

#define BS   65536
#define HID  256
#define NQ   8
#define NV   8
#define NL   36

// ---------------- workspace layout (dwords) ----------------
// W2P  : 8 kc * 16 nt * 256 = 32768   (B frags, fwd  h1@W2)
// W2TP : 32768                        (B frags, bwd  g@W2^T)
// W3P  : 8 kc *  3 nt * 256 = 6144    (B frags, fwd  h2@W3, N padded to 48)
// W3TP : 2 kc * 16 nt * 256 = 8192    (B frags, bwd  g_l@W3^T, K padded 64)
#define WS_W2P  0
#define WS_W2TP 32768
#define WS_W3P  65536
#define WS_W3TP 71680

// Pack an f32 matrix into the CDNA5 WMMA 16-bit B-fragment layout:
// lane n (0-15) holds K = kbase+0..7 and kbase+16..23 of column n; lanes 16-31 hold K+8.
// dword index = ((c*NT + nt)*32 + lane)*8 + v ; halves = K = base+8*(lane>>4)+16*(v>>2)+2*(v&3) and that +1
__global__ void lnn_pack_b(const float* __restrict__ src, uint32_t* __restrict__ dst,
                           int CC, int NT, int Kmax, int Nmax, int sK, int sN) {
  int idx = blockIdx.x * blockDim.x + threadIdx.x;
  int total = CC * NT * 256;
  if (idx >= total) return;
  int v    = idx & 7;
  int lane = (idx >> 3) & 31;
  int rest = idx >> 8;
  int nt   = rest % NT;
  int c    = rest / NT;
  int nn   = nt * 16 + (lane & 15);
  int k0   = c * 32 + ((lane >> 4) & 1) * 8 + ((v >> 2) & 1) * 16 + 2 * (v & 3);
  float a = (k0     < Kmax && nn < Nmax) ? src[(size_t)k0 * sK + (size_t)nn * sN]       : 0.f;
  float b = (k0 + 1 < Kmax && nn < Nmax) ? src[(size_t)(k0 + 1) * sK + (size_t)nn * sN] : 0.f;
  union { _Float16 h[2]; uint32_t u; } cv;
  cv.h[0] = (_Float16)a; cv.h[1] = (_Float16)b;
  dst[idx] = cv.u;
}

// ---- packed f16 A-layout (activation) helpers: 16 rows x 256 K in LDS ----
// A 16-bit 16x32 fragment mapping (ISA 7.12.2): within K-chunk c (32 wide),
// k32 bits: [0]=half, [2:1]=v&3, [3]=lane+16, [4]=v+4.
__device__ __forceinline__ void storeHalfA(uint32_t* smem, int off, int m, int k, float val) {
  _Float16* hp = reinterpret_cast<_Float16*>(smem + off);
  int c = k >> 5, k32 = k & 31;
  int lp = m + (((k32 >> 3) & 1) << 4);
  int v  = (((k32 >> 4) & 1) << 2) + ((k32 >> 1) & 3);
  hp[((((c * 32 + lp) * 8) + v) << 1) + (k32 & 1)] = (_Float16)val;
}
__device__ __forceinline__ float loadHalfA(const uint32_t* smem, int off, int m, int k) {
  const _Float16* hp = reinterpret_cast<const _Float16*>(smem + off);
  int c = k >> 5, k32 = k & 31;
  int lp = m + (((k32 >> 3) & 1) << 4);
  int v  = (((k32 >> 4) & 1) << 2) + ((k32 >> 1) & 3);
  return (float)hp[((((c * 32 + lp) * 8) + v) << 1) + (k32 & 1)];
}

__device__ __forceinline__ float softplus_f(float x) {
  return (x > 20.f) ? x : log1pf(__expf(x));
}

// 4 waves x 16 batch rows = 64 rows/block. Dynamic LDS = 32768 dwords (128KB).
__global__ __launch_bounds__(128) void lnn_main(
    const float* __restrict__ q,  const float* __restrict__ qd, const float* __restrict__ tau,
    const float* __restrict__ W1, const float* __restrict__ b1,
    const float* __restrict__ b2, const float* __restrict__ b3,
    const float* __restrict__ Wv1, const float* __restrict__ bv1, const float* __restrict__ Wv2,
    const uint32_t* __restrict__ W2P, const uint32_t* __restrict__ W2TP,
    const uint32_t* __restrict__ W3P, const uint32_t* __restrict__ W3TP,
    float* __restrict__ out)
{
  extern __shared__ uint32_t smem[];
  const int tid = threadIdx.x;
  const int l   = tid & 31;         // lane
  const int w   = tid >> 5;         // wave in block
  const int n   = l & 15;           // WMMA N / column lane
  const int hi  = l >> 4;           // C/D: rows 0-7 vs 8-15
  const int rowBase = blockIdx.x * 64 + w * 16;

  // LDS layout (dwords)
  const int W1S = 0;                    // 2048: staged W1 (8x256 f32)
  const int DT  = 2048;                 // 512 : dT/dq partials, [wave][16][8]
  const int DV  = 2560;                 // 512 : dV/dq,          [wave][16][8]
  const int PW  = 3072 + w * 7424;      // per-wave block
  const int PH1 = PW;                   // 2048: h1 packed f16 A-layout
  const int PH2 = PW + 2048;            // 2048: h2 packed f16 A-layout
  const int PG  = PW + 4096;            // 512 : g_l packed (K padded to 64)
  const int PG2 = PW + 4608;            // 2048: g_pre2 packed
  const int LSC = PW + 6656;            // 768 : l scratch f32 [16][48]

  float* W1s = reinterpret_cast<float*>(smem + W1S);
  float* dTs = reinterpret_cast<float*>(smem + DT) + w * 128;
  float* dVs = reinterpret_cast<float*>(smem + DV) + w * 128;
  float* lSf = reinterpret_cast<float*>(smem + LSC);

  // ---------- phase 0: stage W1, zero reduction buffers & g_l pad ----------
  for (int i = tid; i < 2048; i += 128) W1s[i] = W1[i];
  for (int i = tid; i < 1024; i += 128) smem[DT + i] = 0u;   // zeros dTs+dVs
  for (int i = l; i < 512; i += 32)     smem[PG + i] = 0u;   // zero-pad g_l K=64
  __syncthreads();

  // ---------- phase 1: h1 = tanh(q@W1+b1), packed f16 into PH1 ----------
  {
    const int m = n;
    const int row = rowBase + m;
    float q8[8];
    #pragma unroll
    for (int j = 0; j < 8; ++j) q8[j] = q[row * 8 + j];
    for (int kp = hi; kp < 128; kp += 2) {       // lanes l and l+16 split k-pairs
      int k = kp * 2;
      float p0 = b1[k], p1 = b1[k + 1];
      #pragma unroll
      for (int j = 0; j < 8; ++j) {
        p0 = fmaf(q8[j], W1s[j * 256 + k],     p0);
        p1 = fmaf(q8[j], W1s[j * 256 + k + 1], p1);
      }
      float t0 = tanhf(p0), t1 = tanhf(p1);
      int k32 = k & 31, c = k >> 5;
      int lp = m + (((k32 >> 3) & 1) << 4);
      int v  = (((k32 >> 4) & 1) << 2) + ((k32 >> 1) & 3);
      union { _Float16 h[2]; uint32_t u; } cv;
      cv.h[0] = (_Float16)t0; cv.h[1] = (_Float16)t1;   // k even -> halves share dword
      smem[PH1 + (c * 32 + lp) * 8 + v] = cv.u;
    }
  }
  __syncthreads();

  // ---------- phase 2: h2 = tanh(h1@W2+b2)  [WMMA, K=256] ----------
  for (int nt = 0; nt < 16; ++nt) {
    float bias = b2[nt * 16 + n];
    v8f acc = {bias, bias, bias, bias, bias, bias, bias, bias};
    #pragma unroll
    for (int c = 0; c < 8; ++c) {
      v16h a = *reinterpret_cast<const v16h*>(smem + PH1 + (c * 32 + l) * 8);
      v16h b = *reinterpret_cast<const v16h*>(W2P + ((c * 16 + nt) * 32 + l) * 8);
      acc = __builtin_amdgcn_wmma_f32_16x16x32_f16(false, a, false, b, (short)0, acc, false, false);
    }
    int k = nt * 16 + n;
    #pragma unroll
    for (int r = 0; r < 8; ++r)
      storeHalfA(smem, PH2, hi * 8 + r, k, tanhf(acc[r]));
  }
  __syncthreads();

  // ---------- phase 3: l = h2@W3 + b3  [WMMA, N padded 36->48] ----------
  for (int nt = 0; nt < 3; ++nt) {
    int col = nt * 16 + n;
    float bias = (col < NL) ? b3[col] : 0.f;
    v8f acc = {bias, bias, bias, bias, bias, bias, bias, bias};
    #pragma unroll
    for (int c = 0; c < 8; ++c) {
      v16h a = *reinterpret_cast<const v16h*>(smem + PH2 + (c * 32 + l) * 8);
      v16h b = *reinterpret_cast<const v16h*>(W3P + ((c * 3 + nt) * 32 + l) * 8);
      acc = __builtin_amdgcn_wmma_f32_16x16x32_f16(false, a, false, b, (short)0, acc, false, false);
    }
    #pragma unroll
    for (int r = 0; r < 8; ++r)
      lSf[(hi * 8 + r) * 48 + col] = acc[r];
  }
  __syncthreads();

  // ---------- phase 4 (divergent): lanes 0-15: L, u, g_l; lanes 16-31: V-net grad --
  float Lmat[36] = {};
  if (l < 16) {
    const int m = l, row = rowBase + m;
    #pragma unroll
    for (int t = 0; t < 36; ++t) Lmat[t] = lSf[m * 48 + t];
    float sig[8];
    #pragma unroll
    for (int i = 0; i < 8; ++i) {
      int di = i * (i + 1) / 2 + i;
      float x = Lmat[di];
      sig[i]   = 1.f / (1.f + __expf(-x));
      Lmat[di] = softplus_f(x) + 1.f;
    }
    float qd8[8];
    #pragma unroll
    for (int j = 0; j < 8; ++j) qd8[j] = qd[row * 8 + j];
    float u[8];
    #pragma unroll
    for (int k = 0; k < 8; ++k) {
      float s = 0.f;
      #pragma unroll
      for (int i = 0; i < 8; ++i)
        if (i >= k) s = fmaf(qd8[i], Lmat[i * (i + 1) / 2 + k], s);
      u[k] = s;
    }
    // dT/dL[i][j] = qd_i * u_j (diag scaled by softplus')
    #pragma unroll
    for (int i = 0; i < 8; ++i)
      #pragma unroll
      for (int j = 0; j <= i; ++j) {
        float g = qd8[i] * u[j];
        if (j == i) g *= sig[i];
        storeHalfA(smem, PG, m, i * (i + 1) / 2 + j, g);
      }
  } else {
    const int mv = l - 16, row = rowBase + mv;
    float q8[8];
    #pragma unroll
    for (int j = 0; j < 8; ++j) q8[j] = q[row * 8 + j];
    float dV[8] = {};
    for (int k = 0; k < 256; ++k) {
      float wv[8];
      #pragma unroll
      for (int j = 0; j < 8; ++j) wv[j] = Wv1[j * 256 + k];
      float pre = bv1[k];
      #pragma unroll
      for (int j = 0; j < 8; ++j) pre = fmaf(q8[j], wv[j], pre);
      float t = tanhf(pre);
      float g = (1.f - t * t) * Wv2[k];
      #pragma unroll
      for (int j = 0; j < 8; ++j) dV[j] = fmaf(g, wv[j], dV[j]);
    }
    #pragma unroll
    for (int j = 0; j < 8; ++j) dVs[mv * 8 + j] = dV[j];
  }
  __syncthreads();

  // ---------- phase 5: g_h2 = g_l@W3^T ; g_pre2 = g_h2*(1-h2^2)  [WMMA K=64] ----
  for (int nt = 0; nt < 16; ++nt) {
    v8f acc = {0.f, 0.f, 0.f, 0.f, 0.f, 0.f, 0.f, 0.f};
    #pragma unroll
    for (int c = 0; c < 2; ++c) {
      v16h a = *reinterpret_cast<const v16h*>(smem + PG + (c * 32 + l) * 8);
      v16h b = *reinterpret_cast<const v16h*>(W3TP + ((c * 16 + nt) * 32 + l) * 8);
      acc = __builtin_amdgcn_wmma_f32_16x16x32_f16(false, a, false, b, (short)0, acc, false, false);
    }
    int k = nt * 16 + n;
    #pragma unroll
    for (int r = 0; r < 8; ++r) {
      int m = hi * 8 + r;
      float h = loadHalfA(smem, PH2, m, k);
      storeHalfA(smem, PG2, m, k, acc[r] * (1.f - h * h));
    }
  }
  __syncthreads();

  // ---------- phase 6: g_h1 = g_pre2@W2^T ; fuse *(1-h1^2) and @W1^T ----------
  float dt[64] = {};
  for (int nt = 0; nt < 16; ++nt) {
    v8f acc = {0.f, 0.f, 0.f, 0.f, 0.f, 0.f, 0.f, 0.f};
    #pragma unroll
    for (int c = 0; c < 8; ++c) {
      v16h a = *reinterpret_cast<const v16h*>(smem + PG2 + (c * 32 + l) * 8);
      v16h b = *reinterpret_cast<const v16h*>(W2TP + ((c * 16 + nt) * 32 + l) * 8);
      acc = __builtin_amdgcn_wmma_f32_16x16x32_f16(false, a, false, b, (short)0, acc, false, false);
    }
    int k = nt * 16 + n;
    float w1c[8];
    #pragma unroll
    for (int j = 0; j < 8; ++j) w1c[j] = W1s[j * 256 + k];
    #pragma unroll
    for (int r = 0; r < 8; ++r) {
      float h  = loadHalfA(smem, PH1, hi * 8 + r, k);
      float gp = acc[r] * (1.f - h * h);
      #pragma unroll
      for (int j = 0; j < 8; ++j) dt[r * 8 + j] = fmaf(gp, w1c[j], dt[r * 8 + j]);
    }
  }
  #pragma unroll
  for (int r = 0; r < 8; ++r) {
    int m = hi * 8 + r;
    #pragma unroll
    for (int j = 0; j < 8; ++j) atomicAdd(&dTs[m * 8 + j], dt[r * 8 + j]);
  }
  __syncthreads();

  // ---------- phase 7: rhs = tau - dT - dV ; M=LL^T solve ; clip ; store ----------
  if (l < 16) {
    const int m = l, row = rowBase + m;
    float rhs[8];
    #pragma unroll
    for (int i = 0; i < 8; ++i)
      rhs[i] = tau[row * 8 + i] - dTs[m * 8 + i] - dVs[m * 8 + i];
    float y[8];
    #pragma unroll
    for (int i = 0; i < 8; ++i) {            // L y = rhs (forward)
      float s = rhs[i];
      #pragma unroll
      for (int j = 0; j < 8; ++j)
        if (j < i) s = fmaf(-Lmat[i * (i + 1) / 2 + j], y[j], s);
      y[i] = s / Lmat[i * (i + 1) / 2 + i];
    }
    float a8[8];
    #pragma unroll
    for (int ii = 7; ii >= 0; --ii) {        // L^T a = y (backward)
      float s = y[ii];
      #pragma unroll
      for (int j = 0; j < 8; ++j)
        if (j > ii) s = fmaf(-Lmat[j * (j + 1) / 2 + ii], a8[j], s);
      a8[ii] = s / Lmat[ii * (ii + 1) / 2 + ii];
    }
    #pragma unroll
    for (int i = 0; i < 8; ++i)
      out[row * 8 + i] = fminf(100.f, fmaxf(-100.f, a8[i]));
  }
}

extern "C" void kernel_launch(void* const* d_in, const int* in_sizes, int n_in,
                              void* d_out, int out_size, void* d_ws, size_t ws_size,
                              hipStream_t stream) {
  (void)in_sizes; (void)n_in; (void)out_size; (void)ws_size;
  const float* q   = (const float*)d_in[0];
  const float* qd  = (const float*)d_in[1];
  const float* tau = (const float*)d_in[2];
  const float* W1  = (const float*)d_in[3];
  const float* b1  = (const float*)d_in[4];
  const float* W2  = (const float*)d_in[5];
  const float* b2  = (const float*)d_in[6];
  const float* W3  = (const float*)d_in[7];
  const float* b3  = (const float*)d_in[8];
  const float* Wv1 = (const float*)d_in[9];
  const float* bv1 = (const float*)d_in[10];
  const float* Wv2 = (const float*)d_in[11];
  // d_in[12] = bv2: constant offset of V, zero gradient -> unused
  float* out = (float*)d_out;
  uint32_t* ws = (uint32_t*)d_ws;

  // pack f16 WMMA B-fragments for the four big GEMMs
  lnn_pack_b<<<(8 * 16 * 256 + 255) / 256, 256, 0, stream>>>(W2, ws + WS_W2P,  8, 16, 256, 256, 256, 1);
  lnn_pack_b<<<(8 * 16 * 256 + 255) / 256, 256, 0, stream>>>(W2, ws + WS_W2TP, 8, 16, 256, 256, 1, 256);
  lnn_pack_b<<<(8 *  3 * 256 + 255) / 256, 256, 0, stream>>>(W3, ws + WS_W3P,  8,  3, 256,  NL, NL, 1);
  lnn_pack_b<<<(2 * 16 * 256 + 255) / 256, 256, 0, stream>>>(W3, ws + WS_W3TP, 2, 16,  NL, 256, 1, NL);

  const int lds_bytes = 32768 * 4;  // 128 KB dynamic LDS (WGP has 320 KB)
  hipFuncSetAttribute((const void*)lnn_main, hipFuncAttributeMaxDynamicSharedMemorySize, lds_bytes);
  lnn_main<<<BS / 64, 128, lds_bytes, stream>>>(q, qd, tau, W1, b1, b2, b3, Wv1, bv1, Wv2,
                                                ws + WS_W2P, ws + WS_W2TP, ws + WS_W3P, ws + WS_W3TP,
                                                out);
}
